// Attention_33741263077427
// MI455X (gfx1250) — matrix operand order repaired and checked by reading, not compiled, not run on hardware
//
#include <hip/hip_runtime.h>

typedef __attribute__((ext_vector_type(16))) __bf16 v16bf;
typedef __attribute__((ext_vector_type(8)))  __bf16 bf16x8;
typedef __attribute__((ext_vector_type(8)))  float  v8f;
typedef __attribute__((ext_vector_type(4)))  unsigned int u32x4;
typedef __attribute__((ext_vector_type(8)))  int i32x8;
typedef __attribute__((ext_vector_type(4)))  int i32x4;

#define DEVI __device__ __forceinline__

#if defined(__gfx1250__) && __has_builtin(__builtin_amdgcn_tensor_load_to_lds)
#define USE_TDM 1
#else
#define USE_TDM 0
#endif

namespace {
constexpr int   kS     = 32;
constexpr float kScale = 0.03125f;   // (32*32)^-0.5 = 1/32
}

// ---------------------------------------------------------------------------
// WMMA fragment loaders (bf16, 16x16x32). Source data must be "index-major,
// K-contiguous": A tile stored as M x K row-major, B tile stored as N x K
// row-major (i.e. B^T).  Layouts per CDNA5 ISA 7.12.2:
//   A (16x32, 16-bit):  lane half h=lane>>4: K = 8h+{0..7} then 16+8h+{0..7}
//   B (32x16, 16-bit):  lane half h=lane>>4: K = 8h+{0..7} then 16+8h+{0..7}
// ---------------------------------------------------------------------------
DEVI v16bf frag_a(const __bf16* base, int stride) {
  const int lane = threadIdx.x & 31;
  const __bf16* p = base + (size_t)(lane & 15) * stride + ((lane >> 4) << 3);
  bf16x8 lo = *(const bf16x8*)(p);
  bf16x8 hi = *(const bf16x8*)(p + 16);
  v16bf f;
#pragma unroll
  for (int i = 0; i < 8; ++i) { f[i] = lo[i]; f[i + 8] = hi[i]; }
  return f;
}

DEVI v16bf frag_b(const __bf16* base, int stride) {
  const int lane = threadIdx.x & 31;
  const __bf16* p = base + (size_t)(lane & 15) * stride + ((lane >> 4) << 3);
  bf16x8 lo = *(const bf16x8*)(p);
  bf16x8 hi = *(const bf16x8*)(p + 16);
  v16bf f;
#pragma unroll
  for (int i = 0; i < 8; ++i) { f[i] = lo[i]; f[i + 8] = hi[i]; }
  return f;
}

DEVI v8f wmma_bf16(v16bf a, v16bf b, v8f c) {
  return __builtin_amdgcn_wmma_f32_16x16x32_bf16(false, a, false, b,
                                                 (short)0, c, false, false);
}

#if USE_TDM
// LDS byte offset of a __shared__ object (generic -> addrspace(3) -> int)
DEVI unsigned lds_off(const void* p) {
  return (unsigned)(unsigned long long)
      (__attribute__((address_space(3))) const char*)p;
}

// Issue a TDM 2D tile load: rows of 32 bf16, `rows` rows, row stride
// `stride` elems, from `gsrc` into LDS offset `lds_byte`.
DEVI void tdm_load_tile(const __bf16* gsrc, unsigned lds_byte, int rows,
                        int stride) {
  unsigned long long ga = (unsigned long long)(size_t)gsrc;
  u32x4 g0;
  g0[0] = 1u;                                   // count=1, user descriptor
  g0[1] = lds_byte;                             // lds_addr
  g0[2] = (unsigned)(ga & 0xffffffffu);         // global_addr[31:0]
  g0[3] = (unsigned)((ga >> 32) & 0x01ffffffu)  // global_addr[56:32]
          | (2u << 30);                         // type = 2 ("image")
  i32x8 g1;
  g1[0] = 0x10000;                   // data_size=1 -> 2 bytes/elem
  g1[1] = (int)(32u << 16);          // tensor_dim0 = 32 (elems per row)
  g1[2] = (int)((unsigned)rows << 16); // tensor_dim1 = rows
  g1[3] = (int)(32u << 16);          // tile_dim0 = 32
  g1[4] = rows;                      // tile_dim1 = rows, tile_dim2 = 0
  g1[5] = stride;                    // tensor_dim0_stride (elems)
  g1[6] = 0;
  g1[7] = 0;
  i32x4 gz;
  gz[0] = 0; gz[1] = 0; gz[2] = 0; gz[3] = 0;
  i32x8 gz8;
#pragma unroll
  for (int i = 0; i < 8; ++i) gz8[i] = 0;
  // 6-arg toolchain form: (g0, g1, g2, g3, <extra group, zero-filled>, cpol)
  __builtin_amdgcn_tensor_load_to_lds(g0, g1, gz, gz, gz8, 0);
}
#endif

// ---------------------------------------------------------------------------
// Conversion kernels
// ---------------------------------------------------------------------------
__global__ __launch_bounds__(256) void cvt_x_kernel(const float* __restrict__ x,
                                                    __bf16* __restrict__ xb, int n) {
  int i = blockIdx.x * 256 + threadIdx.x;
  if (i < n) xb[i] = (__bf16)x[i];
}

// src: (CO, CI, 1, 3, 3) f32  ->  dst: [tap][CO][CI] bf16
__global__ __launch_bounds__(256) void cvt_w_kernel(const float* __restrict__ w,
                                                    __bf16* __restrict__ wr,
                                                    int CO, int CI) {
  int i = blockIdx.x * 256 + threadIdx.x;
  int n = CO * CI * 9;
  if (i >= n) return;
  int c = i % CI;
  int t = i / CI;
  int m = t % CO;
  int tap = t / CO;
  wr[i] = (__bf16)w[((size_t)m * CI + c) * 9 + tap];
}

// ---------------------------------------------------------------------------
// Implicit-GEMM 3x3 conv (per-frame, temporal kernel = 1).
// Block = one (b, s, row-pair): N = 64 pixels, M = CO, K = 9*CI.
// 256 threads (8 waves).  Weight K-chunks are streamed into a
// double-buffered LDS tile by the Tensor Data Mover (wave 0 issues,
// TENSORcnt + workgroup barriers synchronize).
// ---------------------------------------------------------------------------
template <int CI, int CO, int NSETS, bool OUT_F32>
__global__ __launch_bounds__(256) void conv3x3_kernel(
    const __bf16* __restrict__ xin,                        // (B, CI, S, 32, 32)
    const __bf16* __restrict__ w0, const __bf16* __restrict__ w1,
    const __bf16* __restrict__ w2,                         // [9][CO][CI]
    const float* __restrict__ bias0, const float* __restrict__ bias1,
    const float* __restrict__ bias2,
    __bf16* __restrict__ o0, __bf16* __restrict__ o1,
    __bf16* __restrict__ o2,                               // (B, CO, S, 1024) bf16
    float* __restrict__ of)                                // f32 out if OUT_F32
{
  constexpr int MT   = CO / 16;      // m-tiles
  constexpr int NTW  = MT / 2;       // n-tiles per wave
  constexpr int CGS  = CI / 32;      // K-chunks per tap
  constexpr int CPS  = 9 * CGS;      // chunks per weight set
  constexpr int TOT  = NSETS * CPS;  // total K-chunks

  __shared__ __bf16 xs[4][34][CI];   // padded input slab, channel-contiguous
  __shared__ __bf16 wbuf[2][CO][32]; // double-buffered weight K-chunk

  const int bid = blockIdx.x;
  const int yp = bid & 15;           // row-pair
  const int s  = (bid >> 4) & 31;
  const int b  = bid >> 9;
  const int y0 = yp * 2;

  const int wave   = threadIdx.x >> 5;
  const int mstrip = wave % MT;
  const int nbase  = (wave / MT) * NTW;
  const int lane   = threadIdx.x & 31;
  const int ncol   = lane & 15;
  const int rbase  = (lane >> 4) * 8;

  const __bf16* wsets[3] = {w0, w1, w2};
  const float*  bsets[3] = {bias0, bias1, bias2};
  __bf16*       osets[3] = {o0, o1, o2};

#if USE_TDM
  auto issue_chunk = [&](int qc) {
    const int cset = qc / CPS;
    const int rem  = qc % CPS;
    const int ctap = rem / CGS;
    const int ccg  = rem % CGS;
    const __bf16* gsrc = wsets[cset] + (size_t)ctap * CO * CI + ccg * 32;
    tdm_load_tile(gsrc, lds_off(&wbuf[qc & 1][0][0]), CO, CI);
  };
  if (wave == 0) issue_chunk(0);
#endif

  // ---- stage padded input rows y0-1 .. y0+2 into LDS -----------------------
  for (int p = threadIdx.x; p < 4 * CI; p += 256) {
    const int r = p / CI, c = p % CI;
    const int y = y0 - 1 + r;
    const bool valid = (y >= 0) && (y < 32);
    xs[r][0][c]  = (__bf16)0.0f;
    xs[r][33][c] = (__bf16)0.0f;
    if (valid) {
      const __bf16* src = xin + (((size_t)b * CI + c) * kS + s) * 1024 + y * 32;
#pragma unroll
      for (int g = 0; g < 4; ++g) {
        bf16x8 vals = *(const bf16x8*)(src + g * 8);
#pragma unroll
        for (int e = 0; e < 8; ++e) xs[r][1 + g * 8 + e][c] = vals[e];
      }
    } else {
#pragma unroll
      for (int xc = 0; xc < 32; ++xc) xs[r][1 + xc][c] = (__bf16)0.0f;
    }
  }
  __syncthreads();

  int q = 0;
  for (int set = 0; set < NSETS; ++set) {
    v8f zero = {};
    v8f acc[NTW];
#pragma unroll
    for (int j = 0; j < NTW; ++j) acc[j] = zero;

    for (int tap = 0; tap < 9; ++tap) {
      const int dy = tap / 3, dx = tap % 3;
      for (int cg = 0; cg < CGS; ++cg, ++q) {
        v16bf af;
#if USE_TDM
        if (wave == 0) {
          if (q + 1 < TOT) {
            issue_chunk(q + 1);
            __builtin_amdgcn_s_wait_tensorcnt(1);   // chunk q complete
          } else {
            __builtin_amdgcn_s_wait_tensorcnt(0);
          }
        }
        __syncthreads();                            // publish chunk q
        af = frag_a(&wbuf[q & 1][mstrip * 16][0], 32);
#else
        af = frag_a(wsets[set] + ((size_t)tap * CO + mstrip * 16) * CI + cg * 32,
                    CI);
#endif
#pragma unroll
        for (int j = 0; j < NTW; ++j) {
          const int n0 = (nbase + j) * 16;   // tile lies within one output row
          v16bf bf = frag_b(&xs[(n0 >> 5) + dy][(n0 & 31) + dx][cg * 32], CI);
          acc[j] = wmma_bf16(af, bf, acc[j]);
        }
#if USE_TDM
        __syncthreads();   // all waves done with wbuf[q&1] before reuse
#endif
      }
    }

    // ---- epilogue: bias add + store (C layout: col=lane&15, row=8*(l>>4)+r)
    const float* bp = bsets[set];
#pragma unroll
    for (int j = 0; j < NTW; ++j) {
      const int n0  = (nbase + j) * 16;
      const int pix = y0 * 32 + n0 + ncol;
#pragma unroll
      for (int r = 0; r < 8; ++r) {
        const int m = mstrip * 16 + rbase + r;
        const float val = acc[j][r] + bp[m];
        const size_t oidx = (((size_t)b * CO + m) * kS + s) * 1024 + pix;
        if constexpr (OUT_F32) of[oidx] = val;
        else                   osets[set][oidx] = (__bf16)val;
      }
    }
  }
}

// ---------------------------------------------------------------------------
// Temporal attention per (b, c):  attn = sigmoid(Q K^T * scale) ;  O = attn V
// Q,K,V,O: 32 x 1024 bf16 row-major.  128 threads = 4 waves.
// ---------------------------------------------------------------------------
__global__ __launch_bounds__(128) void attn_kernel(
    const __bf16* __restrict__ q, const __bf16* __restrict__ k,
    const __bf16* __restrict__ v, __bf16* __restrict__ o)
{
  __shared__ __bf16 abf[32][32];    // attn (s x t) bf16, A-tile staging
  __shared__ __bf16 vt[128][32];    // V^T chunk: [d][t]

  const size_t base = (size_t)blockIdx.x * kS * 1024;  // (b*CATT + c) * 32 * 1024
  const int wave = threadIdx.x >> 5;
  const int lane = threadIdx.x & 31;
  const int ncol = lane & 15;
  const int rbase = (lane >> 4) * 8;

  // ---- GEMM1: one 16x16 attn tile per wave, K = 1024 ----------------------
  const int m0 = (wave & 1) * 16;
  const int n0 = (wave >> 1) * 16;
  v8f acc = {};
  for (int kc = 0; kc < 1024; kc += 32) {
    v16bf aq = frag_a(q + base + (size_t)m0 * 1024 + kc, 1024);
    v16bf bk = frag_b(k + base + (size_t)n0 * 1024 + kc, 1024);
    acc = wmma_bf16(aq, bk, acc);
  }
#pragma unroll
  for (int r = 0; r < 8; ++r) {
    const float sg = 1.0f / (1.0f + __expf(-(acc[r] * kScale)));
    abf[m0 + rbase + r][n0 + ncol] = (__bf16)sg;
  }
  __syncthreads();

  // attn A-fragment (stable for the rest of the kernel)
  v16bf af = frag_a(&abf[(wave & 1) * 16][0], 32);

  // ---- GEMM2: O = attn * V, processed in 128-wide d-chunks ----------------
  for (int dc = 0; dc < 1024; dc += 128) {
    __syncthreads();   // all waves done reading previous vt chunk
    for (int idx = threadIdx.x; idx < 512; idx += 128) {
      const int t = idx >> 4, dg = idx & 15;
      bf16x8 vals = *(const bf16x8*)(v + base + (size_t)t * 1024 + dc + dg * 8);
#pragma unroll
      for (int e = 0; e < 8; ++e) vt[dg * 8 + e][t] = vals[e];
    }
    __syncthreads();

#pragma unroll
    for (int j = 0; j < 4; ++j) {
      const int dn0 = (wave >> 1) * 64 + j * 16;
      v16bf bf = frag_b(&vt[dn0][0], 32);
      v8f oc = {};
      oc = wmma_bf16(af, bf, oc);
#pragma unroll
      for (int r = 0; r < 8; ++r) {
        const int srow = (wave & 1) * 16 + rbase + r;
        o[base + (size_t)srow * 1024 + dc + dn0 + ncol] = (__bf16)oc[r];
      }
    }
  }
}

// ---------------------------------------------------------------------------
// Host launcher
// ---------------------------------------------------------------------------
extern "C" void kernel_launch(void* const* d_in, const int* in_sizes, int n_in,
                              void* d_out, int out_size, void* d_ws, size_t ws_size,
                              hipStream_t stream) {
  (void)in_sizes; (void)n_in; (void)out_size; (void)ws_size;

  const float* x  = (const float*)d_in[0];
  const float* Wq = (const float*)d_in[1];
  const float* bq = (const float*)d_in[2];
  const float* Wk = (const float*)d_in[3];
  const float* bk = (const float*)d_in[4];
  const float* Wv = (const float*)d_in[5];
  const float* bv = (const float*)d_in[6];
  const float* Wo = (const float*)d_in[7];
  const float* bo = (const float*)d_in[8];

  constexpr size_t N_X   = (size_t)16 * 64 * 32 * 1024;    // 33,554,432
  constexpr size_t N_QKV = (size_t)16 * 128 * 32 * 1024;   // 67,108,864
  constexpr size_t N_W   = (size_t)9 * 128 * 64;           // 73,728

  char* ws = (char*)d_ws;
  size_t off = 0;
  __bf16* xb = (__bf16*)(ws + off); off += N_X * 2;
  __bf16* qb = (__bf16*)(ws + off); off += N_QKV * 2;
  __bf16* kb = (__bf16*)(ws + off); off += N_QKV * 2;
  __bf16* vb = (__bf16*)(ws + off); off += N_QKV * 2;
  __bf16* ob = (__bf16*)(ws + off); off += N_QKV * 2;
  __bf16* wqr = (__bf16*)(ws + off); off += N_W * 2;
  __bf16* wkr = (__bf16*)(ws + off); off += N_W * 2;
  __bf16* wvr = (__bf16*)(ws + off); off += N_W * 2;
  __bf16* wor = (__bf16*)(ws + off); off += N_W * 2;

  // 1) convert inputs/weights to bf16 (weights reordered to [tap][CO][CI])
  cvt_x_kernel<<<(int)(N_X / 256), 256, 0, stream>>>(x, xb, (int)N_X);
  const int wblk = (int)((N_W + 255) / 256);
  cvt_w_kernel<<<wblk, 256, 0, stream>>>(Wq, wqr, 128, 64);
  cvt_w_kernel<<<wblk, 256, 0, stream>>>(Wk, wkr, 128, 64);
  cvt_w_kernel<<<wblk, 256, 0, stream>>>(Wv, wvr, 128, 64);
  cvt_w_kernel<<<wblk, 256, 0, stream>>>(Wo, wor, 64, 128);

  // 2) fused q/k/v conv  (B*S*16 row-pair blocks)
  conv3x3_kernel<64, 128, 3, false><<<16 * 32 * 16, 256, 0, stream>>>(
      xb, wqr, wkr, wvr, bq, bk, bv, qb, kb, vb, nullptr);

  // 3) temporal attention per (b, c)
  attn_kernel<<<16 * 128, 128, 0, stream>>>(qb, kb, vb, ob);

  // 4) output conv -> f32 result
  conv3x3_kernel<128, 64, 1, true><<<16 * 32 * 16, 256, 0, stream>>>(
      ob, wor, nullptr, nullptr, bo, nullptr, nullptr,
      nullptr, nullptr, nullptr, (float*)d_out);
}
